// XORConv2d_25941602467986
// MI455X (gfx1250) — matrix operand order repaired and checked by reading, not compiled, not run on hardware
//
#include <hip/hip_runtime.h>
#include <hip/hip_bf16.h>

// ---------------------------------------------------------------------------
// XOR-Conv (±1-weight 3x3 VALID conv) as implicit GEMM on CDNA5 WMMA.
//   M = C_OUT = 256, N = B*62*62 = 123008, K = C_IN*9 = 1152
// Weights exact in bf16 (±1); activations split bf16 hi+lo -> ~fp32 accuracy:
//   D = A*Bhi + A*Blo, f32 accumulate.
// Block tile 256(M) x 64(N); 8 waves, each owns a 32-row M stripe.
// A tile staged global->LDS with GLOBAL_LOAD_ASYNC_TO_LDS_B128 (ASYNCcnt).
// LDS rows padded to 40 ushorts (80B = 20-bank stride) -> conflict-free frags.
// d_ws requirement: 36*256*32*2 = 589,824 bytes (packed bf16 weights).
// ---------------------------------------------------------------------------

typedef __attribute__((ext_vector_type(16))) __bf16 v16bf;
typedef __attribute__((ext_vector_type(8)))  float  v8f;
typedef __attribute__((ext_vector_type(4)))  unsigned int u32x4;
typedef __attribute__((ext_vector_type(4)))  int v4i;

union FragAB { v16bf v; u32x4 q[2]; };

#define LDS_STRIDE 40u   // ushorts per 32-element row (8-ushort pad)

#if defined(__HIP_DEVICE_COMPILE__) && __has_builtin(__builtin_amdgcn_global_load_async_to_lds_b128)
#define USE_ASYNC_LDS 1
#else
#define USE_ASYNC_LDS 0
#endif

__device__ __forceinline__ unsigned short bf16_rn(float f) {
    unsigned int u = __float_as_uint(f);
    unsigned int r = 0x7FFFu + ((u >> 16) & 1u);
    return (unsigned short)((u + r) >> 16);
}
__device__ __forceinline__ float bf16_to_f32(unsigned short h) {
    return __uint_as_float(((unsigned int)h) << 16);
}

// ---- pack W (f32 {0,1}) -> eff_w (bf16 {+1,-1}) tiled [kc=36][m=256][k=32] --
// K index = (kh*3+kw)*128 + ci ;  kc = kIdx/32, kin = kIdx%32
__global__ __launch_bounds__(256) void xorconv_pack_w(
        const float* __restrict__ W, unsigned short* __restrict__ Wp) {
    unsigned int idx  = blockIdx.x * 256u + threadIdx.x;   // 256*1152 total
    unsigned int co   = idx / 1152u;
    unsigned int kIdx = idx % 1152u;
    unsigned int pos  = kIdx >> 7;        // kh*3+kw  (0..8)
    unsigned int ci   = kIdx & 127u;
    unsigned int kh   = pos / 3u;
    unsigned int kw   = pos % 3u;
    float w   = W[((co * 128u + ci) * 3u + kh) * 3u + kw];
    float eff = 1.0f - 2.0f * w;          // exactly +-1
    unsigned int kc  = pos * 4u + (ci >> 5);
    unsigned int kin = ci & 31u;
    Wp[(kc * 256u + co) * 32u + kin] = bf16_rn(eff);
}

// ---- implicit-GEMM main kernel --------------------------------------------
__global__ __launch_bounds__(256) void xorconv_wmma_gemm(
        const float* __restrict__ x,
        const unsigned short* __restrict__ Wp,
        float* __restrict__ out) {
    __shared__ unsigned short sA [256 * LDS_STRIDE];   // [m][k] 20 KB
    __shared__ unsigned short sBh[ 64 * LDS_STRIDE];   // [n][k]  5 KB (hi)
    __shared__ unsigned short sBl[ 64 * LDS_STRIDE];   // [n][k]  5 KB (lo)

    const unsigned int tid    = threadIdx.x;
    const unsigned int lane   = tid & 31u;
    const unsigned int wave   = tid >> 5;
    const unsigned int nBlock = blockIdx.x * 64u;    // 1922 blocks, no tail

    // ---- per-thread B-stage pixel (fixed across K loop) ----
    const unsigned int nStage = nBlock + (tid & 63u);
    const unsigned int kq     = tid >> 6;            // k-quarter 0..3
    const unsigned int bImg   = nStage / 3844u;      // 62*62
    const unsigned int rem    = nStage % 3844u;
    const unsigned int ho     = rem / 62u;
    const unsigned int wo     = rem % 62u;
    const unsigned int xBase0 = bImg * 524288u;      // b*128*64*64
    const unsigned int sOffB  = (tid & 63u) * LDS_STRIDE + kq * 8u;

    v8f acc[2][4] = {};                    // [mTile][nTile], 64 VGPRs

    // ---- fragment geometry (ISA 7.12.2) ----
    const unsigned int aRow = lane & 15u;
    const unsigned int aOff = (lane < 16u) ? 0u : 8u;    // A K-half
    const unsigned int bOff = (lane < 16u) ? 0u : 8u;    // B K-half

    for (unsigned int kc = 0; kc < 36u; ++kc) {
        const unsigned int pos    = kc >> 2;          // tap index kh*3+kw
        const unsigned int kh     = pos / 3u;
        const unsigned int kw     = pos % 3u;
        const unsigned int ciBase = (kc & 3u) << 5;

        // -- stage A: 16KB slab, async DMA global->LDS (per 16B chunk) --
        {
            const unsigned short* slab = Wp + kc * 8192u;   // [256][32]
#if USE_ASYNC_LDS
            #pragma unroll
            for (unsigned int p = 0; p < 4u; ++p) {
                const unsigned int i = p * 256u + tid;      // 1024 chunks
                const unsigned int m = i >> 2;
                const unsigned int c = i & 3u;
                __builtin_amdgcn_global_load_async_to_lds_b128(
                    (__attribute__((address_space(1))) v4i*)
                        (slab + m * 32u + c * 8u),
                    (__attribute__((address_space(3))) v4i*)
                        (sA + m * LDS_STRIDE + c * 8u),
                    0, 0);
            }
#else
            #pragma unroll
            for (unsigned int p = 0; p < 4u; ++p) {
                const unsigned int i = p * 256u + tid;
                const unsigned int m = i >> 2;
                const unsigned int c = i & 3u;
                *(u32x4*)(sA + m * LDS_STRIDE + c * 8u) =
                    *(const u32x4*)(slab + m * 32u + c * 8u);
            }
#endif
        }
        // -- stage B: load x f32 (wo-contiguous lanes), split hi/lo bf16 --
        // overlaps with the async A DMA above
        {
            const unsigned int xBase =
                xBase0 + (ho + kh) * 64u + (wo + kw) + (ciBase + kq * 8u) * 4096u;
            #pragma unroll
            for (unsigned int j = 0; j < 8u; ++j) {
                float v = x[xBase + j * 4096u];
                unsigned short hi = bf16_rn(v);
                unsigned short lo = bf16_rn(v - bf16_to_f32(hi));
                sBh[sOffB + j] = hi;
                sBl[sOffB + j] = lo;
            }
        }
#if USE_ASYNC_LDS
#if __has_builtin(__builtin_amdgcn_s_wait_asynccnt)
        __builtin_amdgcn_s_wait_asynccnt(0);
#else
        asm volatile("s_wait_asynccnt 0" ::: "memory");
#endif
#endif
        __syncthreads();

        // -- A fragments: two 16-row M tiles of this wave's 32-row stripe --
        FragAB aF[2];
        #pragma unroll
        for (unsigned int mt = 0; mt < 2u; ++mt) {
            const unsigned short* aPtr =
                sA + (wave * 32u + mt * 16u + aRow) * LDS_STRIDE + aOff;
            aF[mt].q[0] = *(const u32x4*)(aPtr);
            aF[mt].q[1] = *(const u32x4*)(aPtr + 16u);
        }

        #pragma unroll
        for (unsigned int t = 0; t < 4u; ++t) {
            const unsigned int bRow =
                (t * 16u + (lane & 15u)) * LDS_STRIDE + bOff;
            FragAB bH, bL;
            bH.q[0] = *(const u32x4*)(sBh + bRow);
            bH.q[1] = *(const u32x4*)(sBh + bRow + 16u);
            bL.q[0] = *(const u32x4*)(sBl + bRow);
            bL.q[1] = *(const u32x4*)(sBl + bRow + 16u);
            #pragma unroll
            for (unsigned int mt = 0; mt < 2u; ++mt) {
                acc[mt][t] = __builtin_amdgcn_wmma_f32_16x16x32_bf16(
                    false, aF[mt].v, false, bH.v, (short)0, acc[mt][t], false, false);
                acc[mt][t] = __builtin_amdgcn_wmma_f32_16x16x32_bf16(
                    false, aF[mt].v, false, bL.v, (short)0, acc[mt][t], false, false);
            }
        }
        __syncthreads();
    }

    // ---- store D: lane L, vgpr r -> m=(L>>4)*8+r, n=L&15 ----
    #pragma unroll
    for (unsigned int t = 0; t < 4u; ++t) {
        const unsigned int nG = nBlock + t * 16u + (lane & 15u);
        const unsigned int b  = nG / 3844u;
        const unsigned int rr = nG % 3844u;
        const unsigned int oh = rr / 62u;
        const unsigned int ow = rr % 62u;
        const unsigned int oBase = (b * 256u * 62u + oh) * 62u + ow;
        #pragma unroll
        for (unsigned int mt = 0; mt < 2u; ++mt) {
            const unsigned int co0 = wave * 32u + mt * 16u + (lane >> 4) * 8u;
            #pragma unroll
            for (unsigned int r = 0; r < 8u; ++r)
                out[oBase + (co0 + r) * 3844u] = acc[mt][t][r];
        }
    }
}

extern "C" void kernel_launch(void* const* d_in, const int* in_sizes, int n_in,
                              void* d_out, int out_size, void* d_ws, size_t ws_size,
                              hipStream_t stream) {
    const float* x = (const float*)d_in[0];
    const float* W = (const float*)d_in[1];
    float*       o = (float*)d_out;
    unsigned short* Wp = (unsigned short*)d_ws;      // needs 589,824 B

    // pack weights: 256*1152 elements
    xorconv_pack_w<<<1152, 256, 0, stream>>>(W, Wp);

    // GEMM: N blocks = 123008/64 = 1922 (exact), M = 256 in one block
    xorconv_wmma_gemm<<<1922, 256, 0, stream>>>(x, Wp, o);
}
